// GumbelSlotSelector_87479893885286
// MI455X (gfx1250) — matrix operand order repaired and checked by reading, not compiled, not run on hardware
//
#include <hip/hip_runtime.h>
#include <hip/hip_bf16.h>
#include <math.h>
#include <stdint.h>

typedef __attribute__((ext_vector_type(16))) __bf16 v16bf;
typedef __attribute__((ext_vector_type(8)))  float  v8f;

#define SLOTS_PER_B 64
#define D_DIM 128
#define H_DIM 64
// W1 fragment table: 32 frags (4 k-chunks x 4 n-tiles x {hi,lo}) x 32 lanes x 8 dwords
#define W1_FRAG_DWORDS (32 * 32 * 8)

// ---------------------------------------------------------------------------
// Pre-kernel: build W1 bf16 hi/lo B-fragments in per-lane WMMA register layout.
// B (32x16, KxN) layout: lane l = n-col (l%16), h = l/16; element s -> K = 8h+s (s<8),
// 16+8h+(s-8) (s>=8); packed element pair (s=2d, s=2d+1) in dword d (low half = even s).
// frag id f = c*8 + j*2 + part   (c = k-chunk, j = n-tile, part 0=hi 1=lo)
// ---------------------------------------------------------------------------
__global__ __launch_bounds__(32) void build_w1_frags_kernel(
    const float* __restrict__ W1,      // [128,64]
    uint32_t* __restrict__ frags)      // [32][32][8]
{
  const int f    = blockIdx.x;         // 0..31
  const int part = f & 1;
  const int j    = (f >> 1) & 3;
  const int c    = f >> 3;
  const int lane = threadIdx.x;
  const int k    = c * 32 + (lane >> 4) * 8;

  uint32_t* __restrict__ dst = frags + ((size_t)f * 32 + lane) * 8;
#pragma unroll
  for (int d = 0; d < 8; ++d) {
    const float x0 = W1[(k + 2 * d + (d >= 4 ? 8 : 0) + 0) * H_DIM + j * 16 + (lane & 15)];
    const float x1 = W1[(k + 2 * d + (d >= 4 ? 8 : 0) + 1) * H_DIM + j * 16 + (lane & 15)];
    const __bf16 h0 = (__bf16)x0;
    const __bf16 h1 = (__bf16)x1;
    __bf16 v0, v1;
    if (part == 0) { v0 = h0; v1 = h1; }
    else {
      v0 = (__bf16)(x0 - (float)h0);
      v1 = (__bf16)(x1 - (float)h1);
    }
    const uint32_t w = (uint32_t)__builtin_bit_cast(unsigned short, v0) |
                       ((uint32_t)__builtin_bit_cast(unsigned short, v1) << 16);
    dst[d] = w;
  }
}

// Split floats into bf16 hi + bf16 lo residual (x ~= hi + lo); A-matrix only.
__device__ __forceinline__ void split_bf16_16(const float* __restrict__ x,
                                              v16bf& hi, v16bf& lo) {
#pragma unroll
  for (int e = 0; e < 16; ++e) {
    const float v = x[e];
    const __bf16 h = (__bf16)v;
    hi[e] = h;
    lo[e] = (__bf16)(v - (float)h);
  }
}

__global__ __launch_bounds__(128) void GumbelSlotSelector_kernel(
    const float* __restrict__ slots,     // [B,64,128]
    const float* __restrict__ gumbel_u,  // [B,64,2]
    const float* __restrict__ fix_u,     // [B,64]
    const v16bf* __restrict__ w1frags,   // [32][32] lanes, 32B each
    const float* __restrict__ b1,        // [64]
    const float* __restrict__ W2,        // [64,2]
    const float* __restrict__ b2,        // [2]
    float* __restrict__ out_decision,    // [B,64]
    float* __restrict__ out_keep)        // [B,64]
{
  __shared__ float lds_h[64][65];      // padded: bank = (row + n) % 64, conflict-free
  __shared__ float lds_logit[64][2];
  __shared__ float lds_dec[64];
  __shared__ float lds_key[64];
  __shared__ int   lds_fix[2];         // [0] = needed flag, [1] = argmax index

  const int tid   = threadIdx.x;
  const int lane  = tid & 31;
  const int wave  = tid >> 5;          // 4 waves, one 16-row tile each
  const int b     = blockIdx.x;

  const int m     = lane & 15;         // M within tile (C/D layout: lanes 0-15 / 16-31)
  const int lhalf = (lane >> 4) & 1;

  // ---------- Layer 1: [16,128] x [128,64] via bf16 hi/lo split WMMA ----------
  v8f acc[4];
#pragma unroll
  for (int j = 0; j < 4; ++j) acc[j] = (v8f)(0.0f);

  const float* __restrict__ arow =
      slots + ((size_t)b * SLOTS_PER_B + (size_t)wave * 16 + m) * D_DIM;

#pragma unroll
  for (int c = 0; c < 4; ++c) {
    // A fragment: elems 0-7 -> cols c*32 + lhalf*8 + (0..7),
    //             elems 8-15 -> cols c*32 + 16 + lhalf*8 + (0..7)
    float af[16];
    const int abase = c * 32 + lhalf * 8;
    const float4 a0 = *(const float4*)(arow + abase);
    const float4 a1 = *(const float4*)(arow + abase + 4);
    const float4 a2 = *(const float4*)(arow + abase + 16);
    const float4 a3 = *(const float4*)(arow + abase + 20);
    af[0]  = a0.x; af[1]  = a0.y; af[2]  = a0.z; af[3]  = a0.w;
    af[4]  = a1.x; af[5]  = a1.y; af[6]  = a1.z; af[7]  = a1.w;
    af[8]  = a2.x; af[9]  = a2.y; af[10] = a2.z; af[11] = a2.w;
    af[12] = a3.x; af[13] = a3.y; af[14] = a3.z; af[15] = a3.w;
    v16bf ahi, alo;
    split_bf16_16(af, ahi, alo);

#pragma unroll
    for (int j = 0; j < 4; ++j) {
      // Precomputed W1 fragments: one 32-byte load per operand, L2-resident.
      const int fbase = (c * 4 + j) * 2;
      const v16bf bhi = w1frags[(size_t)(fbase + 0) * 32 + lane];
      const v16bf blo = w1frags[(size_t)(fbase + 1) * 32 + lane];

      acc[j] = __builtin_amdgcn_wmma_f32_16x16x32_bf16(
          false, ahi, false, bhi, (short)0, acc[j], false, false);
      acc[j] = __builtin_amdgcn_wmma_f32_16x16x32_bf16(
          false, ahi, false, blo, (short)0, acc[j], false, false);
      acc[j] = __builtin_amdgcn_wmma_f32_16x16x32_bf16(
          false, alo, false, bhi, (short)0, acc[j], false, false);
    }
  }

  // Bias + ReLU, scatter h tile to LDS.
  // C/D layout: elem r -> M = r + lhalf*8, N = j*16 + m
#pragma unroll
  for (int j = 0; j < 4; ++j) {
    const float bias = b1[j * 16 + m];
#pragma unroll
    for (int r = 0; r < 8; ++r) {
      float v = acc[j][r] + bias;
      v = v > 0.0f ? v : 0.0f;
      lds_h[wave * 16 + r + lhalf * 8][j * 16 + m] = v;
    }
  }
  __syncthreads();

  // ---------- Layer 2: logits[row][t], one (row,t) per thread ----------
  {
    const int row = tid >> 1;
    const int t   = tid & 1;
    float s = b2[t];
#pragma unroll 8
    for (int n = 0; n < H_DIM; ++n)
      s = fmaf(lds_h[row][n], W2[n * 2 + t], s);
    lds_logit[row][t] = s;
  }
  __syncthreads();

  // ---------- Gumbel hard decision + keep_probs ----------
  if (tid < SLOTS_PER_B) {
    const size_t g = (size_t)b * SLOTS_PER_B + tid;
    const float l0 = lds_logit[tid][0];
    const float l1 = lds_logit[tid][1];

    float u0 = gumbel_u[g * 2 + 0];
    float u1 = gumbel_u[g * 2 + 1];
    u0 = fminf(fmaxf(u0, 1e-10f), 1.0f - 1e-7f);
    u1 = fminf(fmaxf(u1, 1e-10f), 1.0f - 1e-7f);
    const float g0 = -logf(-logf(u0));
    const float g1 = -logf(-logf(u1));
    const float z0 = l0 + g0;   // tau = 1.0
    const float z1 = l1 + g1;
    // argmax picks index 0 on ties -> decision is 1 only if z1 strictly greater
    const float dec = (z1 > z0) ? 1.0f : 0.0f;

    lds_dec[tid] = dec;
    lds_key[tid] = (dec != 0.0f) ? -INFINITY : fix_u[g];
    out_keep[g]  = 1.0f / (1.0f + expf(l0 - l1));   // softmax(logits)[...,1]
  }
  __syncthreads();

  // ---------- _ensure_minimum_slots (LOW_BOUND = 1) ----------
  if (tid == 0) {
    int   nact = 0;
    int   best = 0;
    float bk   = lds_key[0];
    for (int i = 0; i < SLOTS_PER_B; ++i) {
      nact += (lds_dec[i] != 0.0f) ? 1 : 0;
      if (lds_key[i] > bk) { bk = lds_key[i]; best = i; }   // strict > keeps first (stable)
    }
    lds_fix[0] = (nact == 0) ? 1 : 0;
    lds_fix[1] = best;
  }
  __syncthreads();

  if (tid < SLOTS_PER_B) {
    float dec = lds_dec[tid];
    if (lds_fix[0] && tid == lds_fix[1]) dec = 1.0f;
    out_decision[(size_t)b * SLOTS_PER_B + tid] = dec;
  }
}

extern "C" void kernel_launch(void* const* d_in, const int* in_sizes, int n_in,
                              void* d_out, int out_size, void* d_ws, size_t ws_size,
                              hipStream_t stream) {
  const float* slots = (const float*)d_in[0];  // [B,64,128]
  const float* gu    = (const float*)d_in[1];  // [B,64,2]
  const float* fu    = (const float*)d_in[2];  // [B,64]
  const float* W1    = (const float*)d_in[3];  // [128,64]
  const float* b1    = (const float*)d_in[4];  // [64]
  const float* W2    = (const float*)d_in[5];  // [64,2]
  const float* b2    = (const float*)d_in[6];  // [2]

  const int BK = in_sizes[2];        // B * K
  const int B  = BK / SLOTS_PER_B;   // one block per batch row

  float* out_decision = (float*)d_out;
  float* out_keep     = (float*)d_out + BK;

  uint32_t* frags = (uint32_t*)d_ws;           // 32 KB fragment table

  build_w1_frags_kernel<<<32, 32, 0, stream>>>(W1, frags);
  GumbelSlotSelector_kernel<<<B, 128, 0, stream>>>(
      slots, gu, fu, (const v16bf*)frags, b1, W2, b2, out_decision, out_keep);
}
